// DualBranchCFCA_46119358824472
// MI455X (gfx1250) — matrix operand order repaired and checked by reading, not compiled, not run on hardware
//
#include <hip/hip_runtime.h>
#include <hip/hip_bf16.h>

typedef __attribute__((ext_vector_type(16))) __bf16 v16bf;
typedef __attribute__((ext_vector_type(8)))  __bf16 v8bf;
typedef __attribute__((ext_vector_type(2)))  __bf16 v2bf;
typedef __attribute__((ext_vector_type(8)))  float  v8f;
typedef __attribute__((ext_vector_type(4)))  int    v4i;

#define BATCH 16
#define CH    512
#define HW    4096
#define HID   32
#define TOPK  153

#if __has_builtin(__builtin_amdgcn_global_load_async_to_lds_b128)
#define HAVE_ASYNC_LDS 1
#else
#define HAVE_ASYNC_LDS 0
#endif

static __device__ inline void async_cp16(const __bf16* g, __bf16* l) {
#if HAVE_ASYNC_LDS
    __builtin_amdgcn_global_load_async_to_lds_b128(
        (__attribute__((address_space(1))) v4i*)g,
        (__attribute__((address_space(3))) v4i*)(unsigned)(uintptr_t)l,
        0, 0);
#else
    *(uint4*)l = *(const uint4*)g;
#endif
}

static __device__ inline void wait_async0() {
#if HAVE_ASYNC_LDS
#if __has_builtin(__builtin_amdgcn_s_wait_asynccnt)
    __builtin_amdgcn_s_wait_asynccnt(0);
#else
    asm volatile("s_wait_asynccnt 0x0" ::: "memory");
#endif
#endif
}

// pack two f32 into one dword of 2 x bf16 (k, k+1) with a single v_cvt_pk_bf16_f32
static __device__ inline unsigned pack_bf16(float lo, float hi) {
#if __has_builtin(__builtin_amdgcn_cvt_pk_bf16_f32)
    v2bf p = __builtin_amdgcn_cvt_pk_bf16_f32(lo, hi);
    return __builtin_bit_cast(unsigned, p);
#else
    unsigned r;
    asm("v_cvt_pk_bf16_f32 %0, %1, %2" : "=v"(r) : "v"(lo), "v"(hi));
    return r;
#endif
}

// ---------------------------------------------------------------- mean kernel
__global__ __launch_bounds__(256)
void k_mean(const float* __restrict__ x_t, const float* __restrict__ x_c,
            float* __restrict__ mean_t, float* __restrict__ mean_c) {
    const int bc = blockIdx.x;                       // b*512 + ch
    const float* x  = blockIdx.y ? x_c : x_t;
    float*      out = blockIdx.y ? mean_c : mean_t;
    const float4* p = (const float4*)(x + (size_t)bc * HW);
    float s = 0.f;
    for (int i = threadIdx.x; i < HW / 4; i += 256) {
        float4 v = p[i];
        s += v.x + v.y + v.z + v.w;
    }
    __shared__ float red[256];
    red[threadIdx.x] = s;
    __syncthreads();
    for (int off = 128; off > 0; off >>= 1) {
        if (threadIdx.x < off) red[threadIdx.x] += red[threadIdx.x + off];
        __syncthreads();
    }
    if (threadIdx.x == 0) out[bc] = red[0] * (1.0f / (float)HW);
}

// ---------------------------------------------------------------- SE MLP
__global__ __launch_bounds__(512)
void k_se(const float* __restrict__ mean_t, const float* __restrict__ mean_c,
          const float* __restrict__ w1_t, const float* __restrict__ b1_t,
          const float* __restrict__ w2_t, const float* __restrict__ b2_t,
          const float* __restrict__ w1_c, const float* __restrict__ b1_c,
          const float* __restrict__ w2_c, const float* __restrict__ b2_c,
          float* __restrict__ att_t, float* __restrict__ att_c) {
    const bool cb = (blockIdx.x == 1);
    const float* mean = cb ? mean_c : mean_t;
    const float* w1   = cb ? w1_c   : w1_t;
    const float* b1   = cb ? b1_c   : b1_t;
    const float* w2   = cb ? w2_c   : w2_t;
    const float* b2   = cb ? b2_c   : b2_t;
    float*       att  = cb ? att_c  : att_t;

    __shared__ float m[CH];
    __shared__ float h[HID];
    const int t = threadIdx.x;
    for (int b = 0; b < BATCH; ++b) {
        m[t] = mean[b * CH + t];
        __syncthreads();
        if (t < HID) {
            float s = b1[t];
            for (int i = 0; i < CH; ++i) s += m[i] * w1[i * HID + t];
            h[t] = fmaxf(s, 0.f);
        }
        __syncthreads();
        float s = b2[t];
        for (int j = 0; j < HID; ++j) s += h[j] * w2[j * CH + t];
        att[b * CH + t] = 1.f / (1.f + __expf(-s));
        __syncthreads();
    }
}

// ---------------------------------------------------------------- top-k mask
__global__ __launch_bounds__(512)
void k_topk(const float* __restrict__ att_t, float* __restrict__ mask_t) {
    __shared__ float a[CH];
    const int b = blockIdx.x, t = threadIdx.x;
    a[t] = att_t[b * CH + t];
    __syncthreads();
    const float v = a[t];
    int rank = 0;
    for (int j = 0; j < CH; ++j) {
        float u = a[j];
        rank += (u > v) || (u == v && j < t);   // jax top_k tie-break: lower idx first
    }
    mask_t[b * CH + t] = (rank < TOPK) ? 1.f : 0.f;
}

// ---------------------------------------------------------------- row softmax -> bf16 W
__global__ __launch_bounds__(256)
void k_softmax(const float* __restrict__ cross, __bf16* __restrict__ Wbf) {
    const int r = blockIdx.x, t = threadIdx.x;
    const float* row = cross + (size_t)r * CH;
    const float v0 = row[t], v1 = row[t + 256];
    __shared__ float red[256];
    red[t] = fmaxf(v0, v1);
    __syncthreads();
    for (int off = 128; off > 0; off >>= 1) {
        if (t < off) red[t] = fmaxf(red[t], red[t + off]);
        __syncthreads();
    }
    const float mx = red[0];
    __syncthreads();
    const float e0 = __expf(v0 - mx), e1 = __expf(v1 - mx);
    red[t] = e0 + e1;
    __syncthreads();
    for (int off = 128; off > 0; off >>= 1) {
        if (t < off) red[t] += red[t + off];
        __syncthreads();
    }
    const float inv = 1.0f / red[0];
    Wbf[(size_t)r * CH + t]       = (__bf16)(e0 * inv);
    Wbf[(size_t)r * CH + t + 256] = (__bf16)(e1 * inv);
}

// ---------------------------------------------------------------- fused GEMM + epilogue
#define MB 128   // out-channel rows per block
#define NT 64    // spatial cols per block
#define KB 32    // K step (bf16 WMMA K)

__global__ __launch_bounds__(256)
void k_main(const float* __restrict__ x_t, const float* __restrict__ x_c,
            const __bf16* __restrict__ Wbf,
            const float* __restrict__ att_t, const float* __restrict__ att_c,
            const float* __restrict__ mask_t, float* __restrict__ out) {
    const int z  = blockIdx.z;
    const int b  = z >> 1;
    const int br = z & 1;                       // 0: out_t = att_t*x_t + mask*(W@x_c)
                                                // 1: out_c = att_c*(x_c + W@x_t)
    const float* Xg = br ? x_t : x_c;           // GEMM operand
    const float* Xe = br ? x_c : x_t;           // elementwise operand
    const int n0 = blockIdx.x * NT;
    const int m0 = blockIdx.y * MB;
    const int tid = threadIdx.x, lane = tid & 31, wave = tid >> 5;
    const int l15 = lane & 15, half = lane >> 4;

    // double-buffered tiles
    __shared__ __align__(16) __bf16 As[2][MB * KB];   // [m][k], 8 KB each
    __shared__ __align__(16) __bf16 Bs[2][NT * KB];   // [n][k], 4 KB each

    v8f acc[4] = {};                                  // wave strip: 16 rows x 64 cols
    const size_t xbase = (size_t)b * CH * HW;

    // per-thread staging coordinates (fixed across K steps)
    const int kp = tid >> 4;        // B: k-pair 0..15  (covers k = 2kp, 2kp+1)
    const int nq = tid & 15;        // B: n-quad 0..15  (covers n = 4nq .. 4nq+3)

    auto stage = [&](int kk0, __bf16* asb, __bf16* bsb) {
        // ---- A tile: 128x32 bf16 (8 KB) via 2 async 16B chunks per thread
        {
            int c = tid;
            int r = c >> 2, seg = (c & 3) * 8;
            async_cp16(&Wbf[(size_t)(m0 + r) * CH + kk0 + seg], &asb[r * KB + seg]);
            c = tid + 256;
            r = c >> 2; seg = (c & 3) * 8;
            async_cp16(&Wbf[(size_t)(m0 + r) * CH + kk0 + seg], &asb[r * KB + seg]);
        }
        // ---- B tile: transpose to [n][k] with f32->bf16 pk pack (one item/thread)
        {
            const size_t gb = xbase + (size_t)(kk0 + 2 * kp) * HW + n0 + nq * 4;
            const float4 va = *(const float4*)&Xg[gb];
            const float4 vb = *(const float4*)&Xg[gb + HW];
            __bf16* brow = &bsb[2 * kp];
            *(unsigned*)&brow[(nq * 4 + 0) * KB] = pack_bf16(va.x, vb.x);
            *(unsigned*)&brow[(nq * 4 + 1) * KB] = pack_bf16(va.y, vb.y);
            *(unsigned*)&brow[(nq * 4 + 2) * KB] = pack_bf16(va.z, vb.z);
            *(unsigned*)&brow[(nq * 4 + 3) * KB] = pack_bf16(va.w, vb.w);
        }
    };

    int buf = 0;
    stage(0, As[0], Bs[0]);
    wait_async0();
    __syncthreads();

#pragma unroll
    for (int k0 = 0; k0 < CH; k0 += KB) {
        const int nxt = k0 + KB;
        if (nxt < CH) stage(nxt, As[buf ^ 1], Bs[buf ^ 1]);

        const __bf16* asb = As[buf];
        const __bf16* bsb = Bs[buf];

        // A fragment: 16x32 bf16; rows = l15, K = half*8+{0..7} and 16+half*8+{0..7}
        const __bf16* ap = &asb[(wave * 16 + l15) * KB + half * 8];
        const v8bf alo = *(const v8bf*)ap;
        const v8bf ahi = *(const v8bf*)(ap + 16);
        const v16bf a = __builtin_shufflevector(alo, ahi,
            0, 1, 2, 3, 4, 5, 6, 7, 8, 9, 10, 11, 12, 13, 14, 15);
#pragma unroll
        for (int nt = 0; nt < 4; ++nt) {
            // B fragment: 32x16 bf16; col = l15, K = half*8+{0..7} and 16+half*8+{0..7}
            const __bf16* bp = &bsb[(nt * 16 + l15) * KB + half * 8];
            const v8bf blo = *(const v8bf*)bp;
            const v8bf bhi = *(const v8bf*)(bp + 16);
            const v16bf bb = __builtin_shufflevector(blo, bhi,
                0, 1, 2, 3, 4, 5, 6, 7, 8, 9, 10, 11, 12, 13, 14, 15);
            acc[nt] = __builtin_amdgcn_wmma_f32_16x16x32_bf16(
                false, a, false, bb, (short)0, acc[nt], false, false);
        }

        if (nxt < CH) wait_async0();
        __syncthreads();
        buf ^= 1;
    }

    // ---- branchless epilogue ----
    // br==0: out_t = att_t[ch]*Xe + mask_t[ch]*g
    // br==1: out_c = att_c[ch]*(Xe + g) = att_c[ch]*Xe + att_c[ch]*g
    const float* attv = br ? att_c : att_t;    // scale on Xe
    const float* sclv = br ? att_c : mask_t;   // scale on GEMM result
    float* obase = out + (br ? (size_t)BATCH * CH * HW : 0);

    // C layout: lanes 0-15 -> M=r, lanes 16-31 -> M=8+r, N=l15
#pragma unroll
    for (int r = 0; r < 8; ++r) {
        const int ch = m0 + wave * 16 + r + half * 8;
        const float s1 = attv[b * CH + ch];
        const float s2 = sclv[b * CH + ch];
        const size_t rowoff = xbase + (size_t)ch * HW + n0 + l15;
#pragma unroll
        for (int nt = 0; nt < 4; ++nt) {
            const size_t off = rowoff + nt * 16;
            obase[off] = s1 * Xe[off] + s2 * acc[nt][r];
        }
    }
}

// ---------------------------------------------------------------- launch
extern "C" void kernel_launch(void* const* d_in, const int* in_sizes, int n_in,
                              void* d_out, int out_size, void* d_ws, size_t ws_size,
                              hipStream_t stream) {
    const float* x_t  = (const float*)d_in[0];
    const float* x_c  = (const float*)d_in[1];
    const float* w1_t = (const float*)d_in[2];
    const float* b1_t = (const float*)d_in[3];
    const float* w2_t = (const float*)d_in[4];
    const float* b2_t = (const float*)d_in[5];
    const float* w1_c = (const float*)d_in[6];
    const float* b1_c = (const float*)d_in[7];
    const float* w2_c = (const float*)d_in[8];
    const float* b2_c = (const float*)d_in[9];
    const float* cross = (const float*)d_in[10];
    float* out = (float*)d_out;

    char* ws = (char*)d_ws;
    const size_t SZ = (size_t)BATCH * CH * sizeof(float);   // 32 KB
    float*  mean_t = (float*)(ws);
    float*  mean_c = (float*)(ws + SZ);
    float*  att_t  = (float*)(ws + 2 * SZ);
    float*  att_c  = (float*)(ws + 3 * SZ);
    float*  mask_t = (float*)(ws + 4 * SZ);
    __bf16* Wbf    = (__bf16*)(ws + 5 * SZ);                // 512 KB

    k_mean<<<dim3(BATCH * CH, 2), 256, 0, stream>>>(x_t, x_c, mean_t, mean_c);
    k_se<<<2, 512, 0, stream>>>(mean_t, mean_c, w1_t, b1_t, w2_t, b2_t,
                                w1_c, b1_c, w2_c, b2_c, att_t, att_c);
    k_topk<<<BATCH, 512, 0, stream>>>(att_t, mask_t);
    k_softmax<<<CH, 256, 0, stream>>>(cross, Wbf);
    k_main<<<dim3(HW / NT, CH / MB, BATCH * 2), 256, 0, stream>>>(
        x_t, x_c, Wbf, att_t, att_c, mask_t, out);
}
